// NonLinearConv2d_31430570672979
// MI455X (gfx1250) — compile-verified, matrix operand order checked by reading, not run on hardware
//
#include <hip/hip_runtime.h>

typedef __attribute__((ext_vector_type(16))) _Float16 v16h;
typedef __attribute__((ext_vector_type(8)))  float    v8f;

#define NB     32      // batch
#define CIN    64
#define HH     32
#define WW     32
#define COUT   128
#define CK     576     // CIN * 3 * 3
#define NCHEB  32      // Chebyshev degree+1 (K per WMMA step)
#define PIX    1024    // H*W
#define MT     128     // pixels per block (M tile)
#define R_DOM      5.0f
#define INV_DENOM  13.3333333333f   // 1/(2*1.5*0.025)
#define SHIFT_Z    1.3333333333f    // 0.1 * INV_DENOM
#define OUT_SCALE  5.625e-5f        // ALPHA * R

__device__ __forceinline__ float softplus_f(float z) {
  return (z > 0.f) ? (z + log1pf(__expf(-z))) : log1pf(__expf(z));
}

// ---------------------------------------------------------------------------
// Kernel A: per-theta Chebyshev coefficients of
//   g(x) = softplus((x-th)*a)^2 - softplus((x-th)*a - s)^2,  x in [-R, R]
// written directly in the v_wmma_f32_16x16x32_f16 B-fragment lane layout:
//   element offset = ((ck*8 + nblk)*32 + lane)*16 + i,  j(lane,i) per ISA table
// ---------------------------------------------------------------------------
__global__ void cheb_coeff_kernel(const float* __restrict__ theta,
                                  _Float16* __restrict__ Bpk) {
  int id = blockIdx.x * blockDim.x + threadIdx.x;
  if (id >= COUT * CK) return;
  int co = id / CK, ck = id % CK;
  float th = theta[(size_t)co * CK + ck];

  float acc[NCHEB];
#pragma unroll
  for (int j = 0; j < NCHEB; ++j) acc[j] = 0.f;

  // Chebyshev-Gauss quadrature at 32 nodes
  for (int m = 0; m < NCHEB; ++m) {
    float t = __cosf(3.14159265358979f * ((float)m + 0.5f) / (float)NCHEB);
    float x = R_DOM * t;
    float z = (x - th) * INV_DENOM;
    float s1 = softplus_f(z);
    float s2 = softplus_f(z - SHIFT_Z);
    float g = s1 * s1 - s2 * s2;
    float Tp = 1.f, Tc = t;
    acc[0] += g;
    acc[1] += g * t;
#pragma unroll
    for (int j = 2; j < NCHEB; ++j) {
      float Tn = 2.f * t * Tc - Tp;
      Tp = Tc; Tc = Tn;
      acc[j] += g * Tn;
    }
  }

  int nblk = co >> 4, n = co & 15;
#pragma unroll
  for (int j = 0; j < NCHEB; ++j) {
    float cj = ((j == 0) ? 1.f : 2.f) * (1.f / (float)NCHEB) * acc[j];
    int L, i;
    if (j < 8)       { L = n;      i = j;      }   // V0-3, lanes 0-15:  K=0..7
    else if (j < 16) { L = n + 16; i = j - 8;  }   // V0-3, lanes 16-31: K=8..15
    else if (j < 24) { L = n;      i = j - 8;  }   // V4-7, lanes 0-15:  K=16..23
    else             { L = n + 16; i = j - 16; }   // V4-7, lanes 16-31: K=24..31
    Bpk[(((size_t)ck * 8 + nblk) * 32 + L) * 16 + i] = (_Float16)cj;
  }
}

// One N-subtile WMMA step; NBI as template parameter so the reuse_a flag is a
// frontend-visible integer constant (set on steps 1..7, which share A with
// the previous identical instruction per ISA §7.12 OPSEL rules).
template <int NBI>
__device__ __forceinline__ void wmma_step(const _Float16* __restrict__ bbase,
                                          const v16h& af, v8f* acc) {
  v16h bf = *(const v16h*)(bbase + NBI * 512);
  acc[NBI] = __builtin_amdgcn_wmma_f32_16x16x32_f16(
      false, af, false, bf, (short)0, acc[NBI], NBI != 0, false);
}

// ---------------------------------------------------------------------------
// Kernel B: fused Chebyshev-basis GEMM.
//   block = 128 pixels (one batch image, 4 rows) x all 128 couts, 8 waves.
//   Per K-step (one patch element ck) each wave reads ONE x value per row
//   from LDS, builds the 16x32 f16 A fragment via the T_j recurrence, and
//   issues 8 v_wmma_f32_16x16x32_f16 (A shared -> reuse_a hint on 1..7).
// ---------------------------------------------------------------------------
__global__ __launch_bounds__(256) void nlconv_wmma_kernel(
    const float* __restrict__ x, const _Float16* __restrict__ Bpk,
    float* __restrict__ out) {
  __shared__ float xs[CIN][6][34];   // 4 output rows + halo, W + halo  (~52 KB)

  int tid = threadIdx.x;
  int m0  = blockIdx.x * MT;
  int b   = m0 / PIX;
  int oy0 = (m0 % PIX) / WW;         // tile covers output rows oy0..oy0+3

  // stage input tile (zero-padded halo)
  for (int idx = tid; idx < CIN * 6 * 34; idx += 256) {
    int c = idx / (6 * 34), rem = idx % (6 * 34);
    int y = rem / 34, xc = rem % 34;
    int iy = oy0 - 1 + y, ix = xc - 1;
    float v = 0.f;
    if (iy >= 0 && iy < HH && ix >= 0 && ix < WW)
      v = x[(((size_t)b * CIN + c) * HH + iy) * WW + ix];
    xs[c][y][xc] = v;
  }
  __syncthreads();

  int wave = tid >> 5, lane = tid & 31;
  int r    = lane & 15;              // A/C row within the wave's 16-row strip
  bool hi  = (lane >= 16);
  int prow = wave * 16 + r;          // pixel-in-tile 0..127
  int oy   = prow >> 5;              // 0..3
  int ox   = prow & 31;

  v8f acc[8];
#pragma unroll
  for (int nb = 0; nb < 8; ++nb) {
#pragma unroll
    for (int e = 0; e < 8; ++e) acc[nb][e] = 0.f;
  }

  for (int ck = 0; ck < CK; ++ck) {
    int c = ck / 9, kk = ck % 9;
    int ky = kk / 3, kx = kk % 3;

    float xv = xs[c][oy + ky][ox + kx];
    float t  = xv * (1.f / R_DOM);
    t = fminf(1.f, fmaxf(-1.f, t));

    // Chebyshev basis T_0..T_31 of this row's x value
    float T[NCHEB];
    T[0] = 1.f; T[1] = t;
#pragma unroll
    for (int j = 2; j < NCHEB; ++j) T[j] = 2.f * t * T[j - 1] - T[j - 2];

    // pack this lane's 16 halves of the 16x32 f16 A fragment (ISA layout)
    v16h af;
#pragma unroll
    for (int i = 0; i < 16; ++i) {
      int j = (i < 8 ? i : i + 8) + (hi ? 8 : 0);
      af[i] = (_Float16)T[j];
    }

    const _Float16* bbase = Bpk + (size_t)ck * 4096 + (size_t)lane * 16;
    if (ck + 1 < CK)
      __builtin_prefetch(bbase + 4096, 0, 1);   // global_prefetch_b8, next K-step

    wmma_step<0>(bbase, af, acc);
    wmma_step<1>(bbase, af, acc);
    wmma_step<2>(bbase, af, acc);
    wmma_step<3>(bbase, af, acc);
    wmma_step<4>(bbase, af, acc);
    wmma_step<5>(bbase, af, acc);
    wmma_step<6>(bbase, af, acc);
    wmma_step<7>(bbase, af, acc);
  }

  // epilogue: acc[nb][vr] maps to 8 CONSECUTIVE pixels -> one 32B store each
  int n = lane & 15;
  int pbase = (m0 % PIX) + wave * 16 + (hi ? 8 : 0);
#pragma unroll
  for (int nb = 0; nb < 8; ++nb) {
    int co = nb * 16 + n;
    v8f o;
#pragma unroll
    for (int vr = 0; vr < 8; ++vr) {
      float v = acc[nb][vr] * OUT_SCALE;
      o[vr] = fminf(9.f, fmaxf(0.f, v));
    }
    *(v8f*)(out + ((size_t)b * COUT + co) * PIX + pbase) = o;
  }
}

extern "C" void kernel_launch(void* const* d_in, const int* in_sizes, int n_in,
                              void* d_out, int out_size, void* d_ws, size_t ws_size,
                              hipStream_t stream) {
  const float* x     = (const float*)d_in[0];   // [32,64,32,32]
  const float* theta = (const float*)d_in[1];   // [128,64,3,3]
  float* out = (float*)d_out;                   // [32,128,32,32]
  _Float16* Bpk = (_Float16*)d_ws;              // 576*8*512 halves = 4.72 MB

  cheb_coeff_kernel<<<(COUT * CK + 255) / 256, 256, 0, stream>>>(theta, Bpk);
  nlconv_wmma_kernel<<<NB * (PIX / MT), 256, 0, stream>>>(x, Bpk, out);
}
